// SparseKernelHmmLm_18897856103323
// MI455X (gfx1250) — matrix operand order repaired and checked by reading, not compiled, not run on hardware
//
#include <hip/hip_runtime.h>

typedef __attribute__((ext_vector_type(16))) _Float16 v16h;
typedef __attribute__((ext_vector_type(8)))  _Float16 v8h;
typedef __attribute__((ext_vector_type(8)))  float    v8f;

#define C_DIM 128
#define D_DIM 64

__global__ __launch_bounds__(256) void logmvv_wmma_kernel(
    const float* __restrict__ A,   // (nt, 128)      log potentials
    const float* __restrict__ B,   // (nt, 128, 64)  left features
    const float* __restrict__ Cc,  // (nt, 128, 64)  right features
    float* __restrict__ Out,       // (nt, 64, 64)
    int nt)
{
    __shared__ float    aS[C_DIM];
    __shared__ float    pm1[16 * D_DIM];
    __shared__ float    pm2[16 * D_DIM];
    __shared__ float    m1s[D_DIM];
    __shared__ float    m2s[D_DIM];
    __shared__ _Float16 E1t[D_DIM * C_DIM];  // transposed: [d][c]
    __shared__ _Float16 E2t[D_DIM * C_DIM];  // transposed: [d][c]

    const int r = blockIdx.x;
    const int t = threadIdx.x;

    const float4* b4 = (const float4*)(B  + (size_t)r * C_DIM * D_DIM);
    const float4* c4 = (const float4*)(Cc + (size_t)r * C_DIM * D_DIM);

    if (t < C_DIM) aS[t] = A[(size_t)r * C_DIM + t];

    // ---- single global pass: stage b,c rows in registers (8 float4 each) ----
    float4 bf[8], cf[8];
#pragma unroll
    for (int q = 0; q < 8; ++q) {
        int idx = t + 256 * q;        // 2048 float4 per row
        bf[q] = b4[idx];
        cf[q] = c4[idx];
    }
    __syncthreads();  // aS visible

    // thread t covers columns jb..jb+3, rows crow = u + 16q (q=0..7)
    const int jb = (t & 15) * 4;
    const int u  = t >> 4;

    float p1[4], p2[4];
#pragma unroll
    for (int j = 0; j < 4; ++j) { p1[j] = -1e30f; p2[j] = -1e30f; }
#pragma unroll
    for (int q = 0; q < 8; ++q) {
        int   crow = u + 16 * q;
        float av   = aS[crow];
        p1[0] = fmaxf(p1[0], av + bf[q].x);
        p1[1] = fmaxf(p1[1], av + bf[q].y);
        p1[2] = fmaxf(p1[2], av + bf[q].z);
        p1[3] = fmaxf(p1[3], av + bf[q].w);
        p2[0] = fmaxf(p2[0], cf[q].x);
        p2[1] = fmaxf(p2[1], cf[q].y);
        p2[2] = fmaxf(p2[2], cf[q].z);
        p2[3] = fmaxf(p2[3], cf[q].w);
    }
#pragma unroll
    for (int j = 0; j < 4; ++j) {
        pm1[u * D_DIM + jb + j] = p1[j];
        pm2[u * D_DIM + jb + j] = p2[j];
    }
    __syncthreads();

    // tree-reduce 16 partials per column -> m1s/m2s
    if (t < D_DIM) {
        float m1v = -1e30f, m2v = -1e30f;
#pragma unroll
        for (int w = 0; w < 16; ++w) {
            m1v = fmaxf(m1v, pm1[w * D_DIM + t]);
            m2v = fmaxf(m2v, pm2[w * D_DIM + t]);
        }
        m1s[t] = m1v;
        m2s[t] = m2v;
    }
    __syncthreads();

    // ---- exp + f16 convert + transposed store to LDS ----
#pragma unroll
    for (int q = 0; q < 8; ++q) {
        int   crow = u + 16 * q;
        float av   = aS[crow];
        E1t[(jb + 0) * C_DIM + crow] = (_Float16)__expf(av + bf[q].x - m1s[jb + 0]);
        E1t[(jb + 1) * C_DIM + crow] = (_Float16)__expf(av + bf[q].y - m1s[jb + 1]);
        E1t[(jb + 2) * C_DIM + crow] = (_Float16)__expf(av + bf[q].z - m1s[jb + 2]);
        E1t[(jb + 3) * C_DIM + crow] = (_Float16)__expf(av + bf[q].w - m1s[jb + 3]);
        E2t[(jb + 0) * C_DIM + crow] = (_Float16)__expf(cf[q].x - m2s[jb + 0]);
        E2t[(jb + 1) * C_DIM + crow] = (_Float16)__expf(cf[q].y - m2s[jb + 1]);
        E2t[(jb + 2) * C_DIM + crow] = (_Float16)__expf(cf[q].z - m2s[jb + 2]);
        E2t[(jb + 3) * C_DIM + crow] = (_Float16)__expf(cf[q].w - m2s[jb + 3]);
    }
    __syncthreads();

    // ---- WMMA phase: out[i,j] = sum_c e1[c,i]*e2[c,j]  (16 tiles, 2 per wave) ----
    const int wave = t >> 5;
    const int lane = t & 31;
    const int l15  = lane & 15;
    const int hi   = lane >> 4;

    float* outRow = Out + (size_t)r * (D_DIM * D_DIM);

#pragma unroll
    for (int tsel = 0; tsel < 2; ++tsel) {
        int tile = wave + tsel * 8;
        int i0   = (tile >> 2) << 4;   // output-row tile base (e1 column)
        int j0   = (tile & 3) << 4;    // output-col tile base (e2 column)

        v8f acc = {};
#pragma unroll
        for (int kc = 0; kc < C_DIM; kc += 32) {
            // A fragment (16x32 f16): lane<16 holds K{p, p+16}; lane>=16 K{p+8, p+24}
            int abase = (i0 + l15) * C_DIM + kc + hi * 8;
            // B fragment (32x16 f16): lane<16 holds K{p, p+16}; lane>=16 K{p+8, p+24}
            int bbase = (j0 + l15) * C_DIM + kc + hi * 8;

            union { v16h v; v8h h[2]; } ua, ub;
            ua.h[0] = *(const v8h*)&E1t[abase];
            ua.h[1] = *(const v8h*)&E1t[abase + 16];
            ub.h[0] = *(const v8h*)&E2t[bbase];
            ub.h[1] = *(const v8h*)&E2t[bbase + 16];

            acc = __builtin_amdgcn_wmma_f32_16x16x32_f16(
                /*neg_a=*/false, ua.v, /*neg_b=*/false, ub.v,
                /*c_mod=*/(short)0, acc, /*reuse_a=*/false, /*reuse_b=*/false);
        }

        // epilogue: C/D layout VGPR v -> M = v + 8*hi, N = l15
#pragma unroll
        for (int v = 0; v < 8; ++v) {
            int   mi  = i0 + v + hi * 8;
            int   nj  = j0 + l15;
            float val = __logf(acc[v]) + m1s[mi] + m2s[nj];
            outRow[mi * D_DIM + nj] = val;
        }
    }
}

extern "C" void kernel_launch(void* const* d_in, const int* in_sizes, int n_in,
                              void* d_out, int out_size, void* d_ws, size_t ws_size,
                              hipStream_t stream)
{
    (void)n_in; (void)out_size; (void)d_ws; (void)ws_size;

    const float* a  = (const float*)d_in[0];
    const float* b  = (const float*)d_in[1];
    const float* c  = (const float*)d_in[2];
    float*       o  = (float*)d_out;

    const int nt = in_sizes[0] / C_DIM;  // N*T real tokens; padding rows never touch d_out

    logmvv_wmma_kernel<<<nt, 256, 0, stream>>>(a, b, c, o, nt);
}
